// CS_19275813225072
// MI455X (gfx1250) — compile-verified, matrix operand order checked by reading, not run on hardware
//
#include <hip/hip_runtime.h>
#include <math.h>

typedef __attribute__((ext_vector_type(16))) _Float16 v16h;
typedef __attribute__((ext_vector_type(8)))  _Float16 v8h;
typedef __attribute__((ext_vector_type(8)))  float    v8f;
typedef unsigned int v4u __attribute__((ext_vector_type(4)));
typedef int          v8i __attribute__((ext_vector_type(8)));
typedef int          v4i __attribute__((ext_vector_type(4)));

#define N_SIG   3072      // B*CH signals
#define NN      2048      // signal length
#define MM      512       // measurements
#define ITERS   100
#define THRESH  0.05f     // STEP * C_L1
#define INV_SCALE 0.01f
#define SST     2064      // padded LDS row stride for S tile (halves): rows 8 banks apart
#define RSTp    528       // padded LDS row stride for R tile (halves): rows 8 banks apart

// workspace layout (bytes)
#define OFF_ETA   0u          // E^T A-swizzled  512x2048 f16 : 2 MB   (GEMM1 A-operand)
#define OFF_EA    2097152u    // E'  A-swizzled 2048x512  f16 : 2 MB   (GEMM2 A-operand)
#define OFF_DA    4194304u    // D^T A-swizzled 2048x2048 f16 : 8 MB   (output A-operand)
#define OFF_BSW   12582912u   // B^T in C-tile order 3072x512 f32 : 6 MB
#define OFF_S     18874368u   // S state 3072x2048 f16 : 12 MB
// total 31457280 bytes (~30 MB), L2-resident

// Orthonormal DCT-II entry: D[k,n] = c_k * cos(pi*k*(2n+1)/(2N)), N=2048.
__device__ __forceinline__ float dct_entry(int k, int two_n_plus_1) {
  int ph = (k * two_n_plus_1) & 8191;                     // exact integer phase mod 2pi
  float c = (k == 0) ? 0.02209708691207961f : 0.03125f;   // rsqrt(2048), sqrt(2/2048)
  return c * __cosf((float)ph * 7.66990393942820655e-4f); // pi/4096
}

__device__ __forceinline__ v8f wmma_f16(v16h a, v16h b, v8f c) {
  return __builtin_amdgcn_wmma_f32_16x16x32_f16(false, a, false, b, (short)0, c,
                                                false, false);
}

__device__ __forceinline__ v16h cat8(v8h lo, v8h hi) {
  return __builtin_shufflevector(lo, hi, 0,1,2,3,4,5,6,7,8,9,10,11,12,13,14,15);
}

__device__ __forceinline__ v8h pack8(v8f x) {
  v8h y;
#pragma unroll
  for (int r = 0; r < 8; ++r) y[r] = (_Float16)x[r];
  return y;
}

// ---- Setup kernels: emit streamed operands directly in WMMA A-fragment order.
// A-operand 16x32 block: lane l holds halves j=0..15 of (m = l&15, K_local),
// K_local = 8*(l>=16) + (j&7) + 16*(j>=8). Block = 512 contiguous halves.

// GEMM1 A: E^T (512 meas x 2048 coeff), blocks (mt in [0,32), kt in [0,64)) at (mt*64+kt)*512
__global__ void build_ETA(_Float16* __restrict__ ETA, const int* __restrict__ idxs) {
  int t = blockIdx.x * 256 + threadIdx.x;           // 2048*512 elements
  int q = t & 511, b = t >> 9;
  int l = q >> 4, j = q & 15;
  int kl = ((l >> 4) << 3) + (j & 7) + ((j >> 3) << 4);
  int mt = b >> 6, kt = b & 63;
  int K = kt * 32 + kl;                             // DCT coefficient 0..2047
  int m = mt * 16 + (l & 15);                       // measurement 0..511
  ETA[t] = (_Float16)dct_entry(K, 2 * idxs[m] + 1); // E^T[m,K] = D[K, idxs[m]]
}

// GEMM2 A: E' (2048 coeff x 512 meas), blocks (mt in [0,128), kt in [0,16)) at (mt*16+kt)*512
__global__ void build_EA(_Float16* __restrict__ EA, const int* __restrict__ idxs) {
  int t = blockIdx.x * 256 + threadIdx.x;           // 2048*512 elements
  int q = t & 511, b = t >> 9;
  int l = q >> 4, j = q & 15;
  int kl = ((l >> 4) << 3) + (j & 7) + ((j >> 3) << 4);
  int mt = b >> 4, kt = b & 15;
  int K = kt * 32 + kl;                             // measurement 0..511
  int n = mt * 16 + (l & 15);                       // coefficient 0..2047
  EA[t] = (_Float16)dct_entry(n, 2 * idxs[K] + 1);  // E'[n,K] = D[n, idxs[K]]
}

// Output A: D^T (2048 n x 2048 k), blocks (mt in [0,128), kt in [0,64)) at (mt*64+kt)*512
__global__ void build_DA(_Float16* __restrict__ DA) {
  int t = blockIdx.x * 256 + threadIdx.x;           // 2048*2048 elements
  int q = t & 511, b = t >> 9;
  int l = q >> 4, j = q & 15;
  int kl = ((l >> 4) << 3) + (j & 7) + ((j >> 3) << 4);
  int mt = b >> 6, kt = b & 63;
  int K = kt * 32 + kl;                             // coefficient k 0..2047
  int n = mt * 16 + (l & 15);                       // sample n 0..2047
  DA[t] = (_Float16)dct_entry(K, 2 * n + 1);        // D^T[n,k] = D[k,n]
}

// B^T (=100*x[...,idxs]^T) in transposed C-tile order: block (rt, mt), 256 floats,
// lane l holds r=0..7 at (M = m = mt*16 + r + 8*(l>=16), N = row = rt*16 + (l&15)).
__global__ void build_Bsw(float* __restrict__ Bsw, const float* __restrict__ x,
                          const int* __restrict__ idxs) {
  int t = blockIdx.x * 256 + threadIdx.x;           // 3072*512 elements
  int q = t & 255, b = t >> 8;
  int l = q >> 3, r = q & 7;
  int rt = b >> 5, mt = b & 31;
  int m   = mt * 16 + r + ((l >> 4) << 3);
  int row = rt * 16 + (l & 15);
  Bsw[t] = 100.0f * x[(size_t)row * NN + idxs[m]];
}

// ---- Persistent ISTA solver: 16 signals per workgroup, 100 iterations in-kernel.
// All GEMMs transposed so the C-tile M-dim is the contiguous dim of S/R:
// vectorized b128 epilogues. S/R/B tiles LDS-resident (~113 KB of 320 KB WGP LDS).
__global__ __launch_bounds__(256, 1)
void ista_solver(_Float16* __restrict__ S,
                 const _Float16* __restrict__ ETA,
                 const _Float16* __restrict__ EA,
                 const float*    __restrict__ Bsw) {
  __shared__ __align__(16) _Float16 Ssh[16 * SST];   // 66,048 B
  __shared__ __align__(16) _Float16 Rsh[16 * RSTp];  // 16,896 B
  __shared__ __align__(16) float    Bsh[32 * 256];   // 32,768 B

  const int tid  = threadIdx.x;
  const int w    = tid >> 5;                        // 8 waves
  const int lane = tid & 31;
  const int mrow = lane & 15;
  const int hi16 = lane >> 4;                       // 0/1
  const int kb   = hi16 << 3;                       // fragment K/element base: 0 or 8
  const int row0 = blockIdx.x << 4;

  // zero the S tile
  for (int i = tid; i < 16 * SST / 2; i += 256) ((unsigned int*)Ssh)[i] = 0u;

  // one-time 32 KB B-tile copy global->LDS via the Tensor Data Mover (wave 0)
#if __has_builtin(__builtin_amdgcn_tensor_load_to_lds)
  if (tid < 32) {
    unsigned long long ga =
        (unsigned long long)(Bsw + (size_t)blockIdx.x * 32 * 256);
    unsigned ldsoff = (unsigned)(unsigned long long)(void*)&Bsh[0];
    v4u g0;
    g0[0] = 1u;                                     // count=1, user descriptor
    g0[1] = ldsoff;                                 // lds_addr (bytes)
    g0[2] = (unsigned)(ga & 0xffffffffu);           // global_addr[31:0]
    g0[3] = (unsigned)((ga >> 32) & 0x01ffffffu) | (2u << 30);  // addr[56:32], type=2
    v8i g1;
    g1[0] = (3 << 16);                              // wg_mask=0, data_size=3 (8 B)
    g1[1] = (4096 << 16);                           // tensor_dim0 = 4096 elems (low 16)
    g1[2] = (1 << 16);                              // tensor_dim0 hi=0, tensor_dim1=1
    g1[3] = (4096 << 16);                           // tensor_dim1 hi=0, tile_dim0=4096
    g1[4] = 0;                                      // tile_dim1/2 unused
    g1[5] = 4096;                                   // tensor_dim0_stride
    g1[6] = 0; g1[7] = 0;
    v4i z4 = {0, 0, 0, 0};
    v8i z8 = {0, 0, 0, 0, 0, 0, 0, 0};
    // 6-arg form on this toolchain: (g0, g1, g2, g3, g4, cpol)
    __builtin_amdgcn_tensor_load_to_lds(g0, g1, z4, z4, z8, 0);
    __builtin_amdgcn_s_wait_tensorcnt(0);
  }
#else
  {
    const float4* __restrict__ src = (const float4*)(Bsw + (size_t)blockIdx.x * 32 * 256);
    float4* dst = (float4*)Bsh;
    for (int i = tid; i < 32 * 256 / 4; i += 256) dst[i] = src[i];
  }
#endif
  __syncthreads();

  const _Float16* Sb = Ssh + mrow * SST + kb;       // S^T B-frag base (LDS)
  const int mt0 = w << 2;                           // GEMM1: 4 m-tiles per wave

#pragma unroll 1
  for (int it = 0; it < ITERS; ++it) {
    // ---------- GEMM1: R^T[512,16] = E^T[512,2048] @ S^T - B^T ----------
    v8f acc[4];
#pragma unroll
    for (int i = 0; i < 4; ++i)
      acc[i] = -*(const v8f*)(Bsh + (mt0 + i) * 256 + lane * 8);

#pragma unroll 4
    for (int kt = 0; kt < NN / 32; ++kt) {
      v16h bS = cat8(*(const v8h*)(Sb + kt * 32), *(const v8h*)(Sb + kt * 32 + 16));
      const _Float16* ab = ETA + ((size_t)(mt0 * 64 + kt) << 9) + lane * 16;
      __builtin_prefetch(ab + 512, 0, 1);           // next K-slab of E^T
#pragma unroll
      for (int i = 0; i < 4; ++i) {
        v16h aE = *(const v16h*)(ab + (size_t)i * 64 * 512);
        acc[i] = wmma_f16(aE, bS, acc[i]);
      }
    }
#pragma unroll
    for (int i = 0; i < 4; ++i)                     // vectorized R store (b128)
      *(v8h*)(&Rsh[mrow * RSTp + (mt0 + i) * 16 + kb]) = pack8(acc[i]);
    __syncthreads();

    // ---------- GEMM2: S = shrink(S - (E' @ R^T)^T) ----------
#pragma unroll 1
    for (int p = 0; p < 2; ++p) {
      v8f g[8] = {};
      const int mtbase = w * 16 + p * 8;            // 8 n-tiles of 16 per pass
#pragma unroll 2
      for (int kt = 0; kt < MM / 32; ++kt) {
        v16h bR = cat8(*(const v8h*)(&Rsh[mrow * RSTp + kt * 32 + kb]),
                       *(const v8h*)(&Rsh[mrow * RSTp + kt * 32 + kb + 16]));
        const _Float16* ab = EA + ((size_t)(mtbase * 16 + kt) << 9) + lane * 16;
        __builtin_prefetch(ab + 512, 0, 1);         // next K-slab of E'
#pragma unroll
        for (int i = 0; i < 8; ++i) {
          v16h aE = *(const v16h*)(ab + (size_t)i * 16 * 512);
          g[i] = wmma_f16(aE, bR, g[i]);
        }
      }
#pragma unroll
      for (int i = 0; i < 8; ++i) {                 // vectorized shrink update (b128)
        int n0 = (mtbase + i) * 16 + kb;
        _Float16* sp = Ssh + mrow * SST + n0;
        v8h sold = *(const v8h*)sp;
        v8h snew;
#pragma unroll
        for (int r = 0; r < 8; ++r) {
          float u = (float)sold[r] - g[i][r];       // s - STEP*grad
          float au = fabsf(u) - THRESH;
          snew[r] = (_Float16)(au > 0.0f ? copysignf(au, u) : 0.0f);
        }
        *(v8h*)sp = snew;
      }
    }
    __syncthreads();   // S/R tiles consistent before next iteration
  }

  // write the converged S tile back to global for the final IDCT GEMM
  for (int v = tid; v < 16 * NN / 8; v += 256) {
    int row = v >> 8, cv = v & 255;                 // 256 v8h per row
    *(v8h*)(S + (size_t)(row0 + row) * NN + cv * 8) =
        *(const v8h*)(Ssh + row * SST + cv * 8);
  }
}

// ---- Final output: out^T = (D^T @ S^T) / SCALE, coalesced v8f stores ---------
__global__ __launch_bounds__(256, 1)
void idct_out(float* __restrict__ out, const _Float16* __restrict__ S,
              const _Float16* __restrict__ DA) {
  const int tid  = threadIdx.x;
  const int w    = tid >> 5, lane = tid & 31;
  const int mrow = lane & 15, hi16 = lane >> 4, kb = hi16 << 3;
  const int row0 = blockIdx.x << 4;
  const _Float16* __restrict__ Sb = S + (size_t)(row0 + mrow) * NN + kb;

#pragma unroll 1
  for (int p = 0; p < 2; ++p) {
    v8f g[8] = {};
    const int mtbase = w * 16 + p * 8;
#pragma unroll 2
    for (int kt = 0; kt < NN / 32; ++kt) {
      v16h bS = cat8(*(const v8h*)(Sb + kt * 32), *(const v8h*)(Sb + kt * 32 + 16));
      const _Float16* ab = DA + ((size_t)(mtbase * 64 + kt) << 9) + lane * 16;
      __builtin_prefetch(ab + 512, 0, 1);
#pragma unroll
      for (int i = 0; i < 8; ++i) {
        v16h aD = *(const v16h*)(ab + (size_t)i * 64 * 512);
        g[i] = wmma_f16(aD, bS, g[i]);
      }
    }
#pragma unroll
    for (int i = 0; i < 8; ++i) {
      int n0 = (mtbase + i) * 16 + kb;
      *(v8f*)(out + (size_t)(row0 + mrow) * NN + n0) = g[i] * INV_SCALE;
    }
  }
}

extern "C" void kernel_launch(void* const* d_in, const int* in_sizes, int n_in,
                              void* d_out, int out_size, void* d_ws, size_t ws_size,
                              hipStream_t stream) {
  (void)in_sizes; (void)n_in; (void)out_size; (void)ws_size;
  const float* x    = (const float*)d_in[0];   // [3072, 2048] fp32
  const int*   idxs = (const int*)d_in[1];     // [512] sorted indices
  char* ws = (char*)d_ws;
  _Float16* ETA = (_Float16*)(ws + OFF_ETA);
  _Float16* EA  = (_Float16*)(ws + OFF_EA);
  _Float16* DA  = (_Float16*)(ws + OFF_DA);
  float*    Bsw = (float*)   (ws + OFF_BSW);
  _Float16* S   = (_Float16*)(ws + OFF_S);
  float*    out = (float*)d_out;

  build_ETA <<< 4096, 256, 0, stream>>>(ETA, idxs);
  build_EA  <<< 4096, 256, 0, stream>>>(EA, idxs);
  build_DA  <<<16384, 256, 0, stream>>>(DA);
  build_Bsw <<< 6144, 256, 0, stream>>>(Bsw, x, idxs);
  ista_solver<<< N_SIG / 16, 256, 0, stream>>>(S, ETA, EA, Bsw);
  idct_out  <<< N_SIG / 16, 256, 0, stream>>>(out, S, DA);
}
